// DotProductAttention_59158879535890
// MI455X (gfx1250) — matrix operand order repaired and checked by reading, not compiled, not run on hardware
//
#include <hip/hip_runtime.h>

typedef _Float16 half_t;
typedef __attribute__((ext_vector_type(16))) _Float16 v16h;
typedef __attribute__((ext_vector_type(8)))  _Float16 v8h;
typedef __attribute__((ext_vector_type(2)))  __fp16   v2hf;   // cvt_pkrtz result type
typedef __attribute__((ext_vector_type(8)))  float    v8f;

#define SEQ   2048
#define BB    2
#define HQN   32
#define HKVN  8
#define DD    128
#define GRP   (HQN / HKVN)        // 4
#define STRK  (BB * HKVN * DD)    // 2048 floats between consecutive s in K/V

// LDS row strides (halves), padded so half-group lanes land on distinct banks
// while keeping every v8h access 16B-aligned:
//  K: 136 halves = 68 dwords -> bank start 4*l16 (each bank hit exactly 2x)
//  V/P: 40 halves = 20 dwords -> bank starts {20*l16 mod 64} all distinct
#define KSTR  136
#define VSTR  40
#define PSTR  40

// scale(1/sqrt(128)) folded with log2(e) and PRE-APPLIED TO Q at load time;
// softmax then runs in the log2 domain via exp2 (v_exp_f32 is base-2 native).
#define SCL2  0.12753102f         // (1/sqrt(128)) * 1.4426950408889634

union PK8 {                       // build a v8h from packed cvt_pkrtz pairs
    v2hf p[4];
    v8h  v;
};
union PK16 {                      // build a v16h from two v8h (no elt moves)
    struct { v8h lo, hi; } h;
    v16h v;
};

__launch_bounds__(256, 2)
__global__ void fa_gqa_causal_kernel(const float* __restrict__ Q,
                                     const float* __restrict__ K,
                                     const float* __restrict__ V,
                                     float* __restrict__ O) {
    // K tile: row-major [key(32)][d(128)+pad] f16  (B-operand of QK^T)
    // V tile: transposed [d(128)][key(32)+pad] f16 (B-operand of P*V)
    // P bounce: per-wave [q(16)][key(32)+pad]      (C-layout -> A-layout)
    __shared__ __align__(32) half_t Klds[32 * KSTR];
    __shared__ __align__(32) half_t Vlds[128 * VSTR];
    __shared__ __align__(32) half_t Plds[8][16 * PSTR];

    const int qb   = blockIdx.x;           // 16 q-blocks of 128 rows
    const int head = blockIdx.y;           // B*HQ = 64
    const int b    = head / HQN;
    const int hq   = head % HQN;
    const int hkv  = hq / GRP;

    const int tid  = threadIdx.x;
    const int wave = tid >> 5;
    const int lane = tid & 31;
    const int l16  = lane & 15;
    const int hi   = lane >> 4;            // half-group: 0 or 1

    const int qbase = qb * 128 + wave * 16;

    const size_t koff0 = ((size_t)b * HKVN + hkv) * DD;   // + key*STRK
    const size_t qrow  = (size_t)(qbase + l16);
    const size_t qoff  = (qrow * BB + b) * HQN * DD + (size_t)hq * DD;

    // ---- Q -> f16 A-layout registers (pre-scaled by SCL2), 4 chunks ----
    // A layout (16-bit, 16x32): v in 0..3 -> K = 2v + 8*hi + {0,1}
    //                           v in 4..7 -> K = 16 + 2(v-4) + 8*hi + {0,1}
    v16h qa[4];
#pragma unroll
    for (int c = 0; c < 4; ++c) {
        const float* qp = Q + qoff + c * 32 + hi * 8;
        PK16 u;
        PK8  a, bb2;
#pragma unroll
        for (int i = 0; i < 4; ++i)
            a.p[i]   = __builtin_amdgcn_cvt_pkrtz(qp[2 * i] * SCL2,      qp[2 * i + 1] * SCL2);
#pragma unroll
        for (int i = 0; i < 4; ++i)
            bb2.p[i] = __builtin_amdgcn_cvt_pkrtz(qp[16 + 2 * i] * SCL2, qp[17 + 2 * i] * SCL2);
        u.h.lo = a.v;
        u.h.hi = bb2.v;
        qa[c] = u.v;
    }

    // constant ones-column B operand: B[key][n] = (n==0); P x B accumulates
    // the row-sums l with the same alpha-recurrence as the O accumulators.
    v16h bones;
    {
        const half_t ov = (l16 == 0) ? (half_t)1.0f : (half_t)0.0f;
#pragma unroll
        for (int i = 0; i < 16; ++i) bones[i] = ov;
    }

    v8f  o[8];                 // O accum, C layout, 8 d-chunks of 16
    v8f  lacc = (v8f){};       // row-sum accum (valid in lanes l16==0)
    float m[8];
#pragma unroll
    for (int dc = 0; dc < 8; ++dc) o[dc] = (v8f){};
#pragma unroll
    for (int r = 0; r < 8; ++r) m[r] = -1e30f;

    // cooperative K mapping: 256 thr, each 16 floats of one key row
    const int ldk = tid >> 3;            // key row 0..31
    const int ldd = (tid & 7) * 16;      // d start 0..112
    // cooperative V mapping: key PAIR x 8 d values -> packed v2hf stores
    const int vkp = (tid & 15) * 2;      // key pair base 0..30
    const int vd8 = (tid >> 4) * 8;      // d start 0..120

    const int nsteps = 4 * (qb + 1);     // uniform causal bound for the block
    const int jmask  = qbase >> 5;       // steps fully below the diagonal

    auto step = [&](const int j, const bool MASKED) {
        const int k0 = j * 32;

        // ---- cooperative K/V tile load + packed fp32->f16 convert ----
        const float* kp  = K + (size_t)(k0 + ldk) * STRK + koff0 + ldd;
        const float* vp0 = V + (size_t)(k0 + vkp) * STRK + koff0 + vd8;
        const float* vp1 = vp0 + STRK;
        {
            PK8 a, bpk;
#pragma unroll
            for (int i = 0; i < 4; ++i) a.p[i]   = __builtin_amdgcn_cvt_pkrtz(kp[2 * i],     kp[2 * i + 1]);
#pragma unroll
            for (int i = 0; i < 4; ++i) bpk.p[i] = __builtin_amdgcn_cvt_pkrtz(kp[8 + 2 * i], kp[9 + 2 * i]);
            *(v8h*)&Klds[ldk * KSTR + ldd]     = a.v;
            *(v8h*)&Klds[ldk * KSTR + ldd + 8] = bpk.v;
#pragma unroll
            for (int i = 0; i < 8; ++i)          // V^T: pair of keys packed
                *(v2hf*)&Vlds[(vd8 + i) * VSTR + vkp] =
                    __builtin_amdgcn_cvt_pkrtz(vp0[i], vp1[i]);
        }
        if (j + 1 < nsteps) {                    // cover HBM/L2 latency
            __builtin_prefetch(kp  + 32 * STRK, 0, 0);
            __builtin_prefetch(vp0 + 32 * STRK, 0, 0);
        }
        __syncthreads();

        // ---- S' = (Q*scl) K^T : preload all 8 B-fragments, then 8 WMMAs ----
        v16h kb[2][4];
#pragma unroll
        for (int t = 0; t < 2; ++t)
#pragma unroll
            for (int c = 0; c < 4; ++c) {
                PK16 u;
                u.h.lo = *(const v8h*)&Klds[(t * 16 + l16) * KSTR + c * 32 + hi * 8];
                u.h.hi = *(const v8h*)&Klds[(t * 16 + l16) * KSTR + c * 32 + 16 + hi * 8];
                kb[t][c] = u.v;
            }
        v8f sacc[2];
#pragma unroll
        for (int t = 0; t < 2; ++t) {
            sacc[t] = (v8f){};
#pragma unroll
            for (int c = 0; c < 4; ++c)
                sacc[t] = __builtin_amdgcn_wmma_f32_16x16x32_f16(
                    false, qa[c], false, kb[t][c], (short)0, sacc[t], false, false);
        }

        // ---- (optional) causal mask + online max (log2 domain) ----
        float p[2][8];
        float rowmax[8];
#pragma unroll
        for (int r = 0; r < 8; ++r) {
            float s0 = sacc[0][r];                   // C layout: M=r+8*hi, N=l16
            float s1 = sacc[1][r];
            if (MASKED) {
                const int row = qbase + r + hi * 8;
                if (k0 + l16 > row)      s0 = -1e30f;
                if (k0 + 16 + l16 > row) s1 = -1e30f;
            }
            p[0][r] = s0; p[1][r] = s1;
            float mx = fmaxf(s0, s1);
#pragma unroll
            for (int d = 1; d < 16; d <<= 1)
                mx = fmaxf(mx, __shfl_xor(mx, d));   // stays in 16-lane half
            rowmax[r] = mx;
        }
        v8f alphav;
#pragma unroll
        for (int r = 0; r < 8; ++r) {
            const float mnew = fmaxf(m[r], rowmax[r]);
            alphav[r] = exp2f(m[r] - mnew);
            m[r] = mnew;
            p[0][r] = exp2f(p[0][r] - mnew);
            p[1][r] = exp2f(p[1][r] - mnew);
        }
#pragma unroll
        for (int dc = 0; dc < 8; ++dc) o[dc] *= alphav;   // v_pk_mul_f32 pairs
        lacc *= alphav;

        // ---- P: C layout -> LDS (row-major [q][key], padded) -> A layout ----
        half_t* Pw = &Plds[wave][0];
#pragma unroll
        for (int t = 0; t < 2; ++t)
#pragma unroll
            for (int i = 0; i < 4; ++i) {
                const v2hf pk = __builtin_amdgcn_cvt_pkrtz(p[t][2 * i], p[t][2 * i + 1]);
                // rows q=2i+8*hi and q=2i+1+8*hi; hi half -> ds_store_b16_d16_hi
                *(__fp16*)&Pw[(2 * i     + 8 * hi) * PSTR + t * 16 + l16] = pk[0];
                *(__fp16*)&Pw[(2 * i + 1 + 8 * hi) * PSTR + t * 16 + l16] = pk[1];
            }
        // same-wave LDS ops are processed in order; only stop compiler reordering
        asm volatile("" ::: "memory");
        v16h pa;
        {
            PK16 u;
            u.h.lo = *(const v8h*)&Pw[l16 * PSTR + hi * 8];
            u.h.hi = *(const v8h*)&Pw[l16 * PSTR + 16 + hi * 8];
            pa = u.v;
        }

        // ---- O += P*V (8 chunks) and l += P*ones (1 extra WMMA) ----
#pragma unroll
        for (int dc = 0; dc < 8; ++dc) {
            PK16 u;
            u.h.lo = *(const v8h*)&Vlds[(dc * 16 + l16) * VSTR + hi * 8];
            u.h.hi = *(const v8h*)&Vlds[(dc * 16 + l16) * VSTR + 16 + hi * 8];
            o[dc] = __builtin_amdgcn_wmma_f32_16x16x32_f16(
                false, pa, false, u.v, (short)0, o[dc], false, false);
        }
        lacc = __builtin_amdgcn_wmma_f32_16x16x32_f16(
            false, pa, false, bones, (short)0, lacc, false, false);
        __syncthreads();   // before next iteration overwrites K/V tiles
    };

    int j = 0;
    for (; j < jmask; ++j)  step(j, false);   // fully causal-valid steps
    for (; j < nsteps; ++j) step(j, true);    // diagonal steps need the mask
    // (barrier counts match across waves: both variants hit 2 barriers/step)

    // ---- epilogue: broadcast l from lane 0/16 of each half, normalize ----
#pragma unroll
    for (int r = 0; r < 8; ++r) {
        const float lsum = __shfl(lacc[r], hi * 16);
        const float inv  = 1.0f / lsum;
        const int   row  = qbase + r + hi * 8;
        float* op = O + ((size_t)row * BB + b) * HQN * DD + (size_t)hq * DD + l16;
#pragma unroll
        for (int dc = 0; dc < 8; ++dc)
            op[dc * 16] = o[dc][r] * inv;
    }
}

extern "C" void kernel_launch(void* const* d_in, const int* in_sizes, int n_in,
                              void* d_out, int out_size, void* d_ws, size_t ws_size,
                              hipStream_t stream) {
    const float* Q = (const float*)d_in[0];
    const float* K = (const float*)d_in[1];
    const float* V = (const float*)d_in[2];
    float* O = (float*)d_out;
    (void)in_sizes; (void)n_in; (void)out_size; (void)d_ws; (void)ws_size;

    dim3 grid(SEQ / 128, BB * HQN);   // (16, 64)
    dim3 block(256);                  // 8 wave32
    hipLaunchKernelGGL(fa_gqa_causal_kernel, grid, block, 0, stream, Q, K, V, O);
}
